// MultiHead_CFAttention_59794534695013
// MI455X (gfx1250) — matrix operand order repaired and checked by reading, not compiled, not run on hardware
//
#include <hip/hip_runtime.h>

typedef __attribute__((ext_vector_type(16))) _Float16 v16h;
typedef __attribute__((ext_vector_type(8)))  float    v8f;

#define WMMA(a, b, c) __builtin_amdgcn_wmma_f32_16x16x32_f16( \
    false, (a), false, (b), (short)0, (c), false, false)

static constexpr int Bz = 4, Nn = 2048, Dd = 512, NHh = 8, HK = 64;
static constexpr int Mrows = Bz * Nn;           // 8192
static constexpr float SCALE = 0.125f;          // 1/sqrt(64)
static constexpr float LN_EPS = 1e-5f;

static __device__ __forceinline__ int lane_id() { return threadIdx.x & 31; }
static __device__ __forceinline__ int wave_id() { return threadIdx.x >> 5; }

// ---- gfx1250 async global->LDS copy (ASYNCcnt-tracked) ---------------------
// Per-lane 16B transfer; LDS dest address in VDST VGPR, global addr in VADDR.
static __device__ __forceinline__ void async_ld_b128(unsigned lds_off,
                                                     const void* gaddr) {
  asm volatile("global_load_async_to_lds_b128 %0, %1, off"
               :: "v"(lds_off), "v"(gaddr)
               : "memory");
}
static __device__ __forceinline__ void wait_async0() {
  asm volatile("s_wait_asynccnt 0" ::: "memory");
}

// ---- WMMA fragment loaders -------------------------------------------------
// A fragment (16x32 f16): lanes 0-15 hold row m=lane, k in {0..7, 16..23};
// lanes 16-31 hold row m=lane-16, k in {8..15, 24..31}.
static __device__ __forceinline__ v16h load_frag_a(const _Float16* base, int ld) {
  const int lane = lane_id();
  const int m = lane & 15;
  const int khi = ((lane >> 4) & 1) << 3;       // 0 or 8
  const _Float16* p = base + (size_t)m * ld;
  v16h a;
#pragma unroll
  for (int i = 0; i < 16; ++i) {
    int k = ((i >> 3) << 4) + khi + (i & 7);
    a[i] = p[k];
  }
  return a;
}

// B fragment (32x16 f16) from a source laid out [col][contraction]
// (contraction contiguous). lanes 0-15: col=lane, k in {0..7, 16..23};
// lanes 16-31: col=lane-16, k in {8..15, 24..31}.
static __device__ __forceinline__ v16h load_frag_b_t(const _Float16* base, int ld) {
  const int lane = lane_id();
  const int n = lane & 15;
  const int k0 = ((lane >> 4) & 1) << 3;        // 0 or 8
  const _Float16* p = base + (size_t)n * ld + k0;
  v16h b;
#pragma unroll
  for (int i = 0; i < 16; ++i) b[i] = p[((i >> 3) << 4) + (i & 7)];
  return b;
}

// C/D fragment: VGPR v -> row = v + 8*(lane>>4), col = lane & 15.

// ---- Kernel 1: projection GEMM  Xf32[8192,512] @ Wf32[512,512] -> f16 ------
// Output written in [b, h, n, hk] layout for the attention kernels.
// Grid: (8192/128, 512/64), block 256 (8 waves); wave w -> 16 rows x 64 cols.
__global__ __launch_bounds__(256) void proj_gemm(const float* __restrict__ X,
                                                 const float* __restrict__ W,
                                                 _Float16* __restrict__ dst) {
  __shared__ _Float16 Xt[128 * 36];   // [row][k], ld 36
  __shared__ _Float16 Wt[64 * 36];    // [col][k], ld 36 (transposed)
  const int tid = threadIdx.x, w = wave_id(), lane = lane_id();
  const int row0 = blockIdx.x * 128;
  const int cb   = blockIdx.y * 64;

  v8f acc[4];
#pragma unroll
  for (int c = 0; c < 4; ++c) acc[c] = (v8f){};

  for (int kb = 0; kb < Dd; kb += 32) {
    __syncthreads();
    { // stage X tile (f32 -> f16), 16 elems / thread
      int r = tid >> 1, c0 = (tid & 1) * 16;
      const float* src = X + (size_t)(row0 + r) * Dd + kb + c0;
#pragma unroll
      for (int i = 0; i < 16; ++i) Xt[r * 36 + c0 + i] = (_Float16)src[i];
    }
    { // stage W tile transposed (f32 -> f16), 8 elems / thread
      int c = tid >> 2, k0 = (tid & 3) * 8;
#pragma unroll
      for (int i = 0; i < 8; ++i)
        Wt[c * 36 + k0 + i] = (_Float16)W[(size_t)(kb + k0 + i) * Dd + cb + c];
      if (kb + 32 < Dd)  // prefetch next weight chunk (global_prefetch_b8)
        __builtin_prefetch(&W[(size_t)(kb + 32 + k0) * Dd + cb + c], 0, 1);
    }
    __syncthreads();
    v16h a = load_frag_a(Xt + w * 16 * 36, 36);
#pragma unroll
    for (int c = 0; c < 4; ++c) {
      v16h b = load_frag_b_t(Wt + c * 16 * 36, 36);
      acc[c] = WMMA(a, b, acc[c]);
    }
  }

  const int g = lane >> 4, nl = lane & 15;
#pragma unroll
  for (int c = 0; c < 4; ++c) {
#pragma unroll
    for (int v = 0; v < 8; ++v) {
      int R = row0 + w * 16 + v + 8 * g;
      int C = cb + c * 16 + nl;
      int b = R >> 11, n = R & (Nn - 1);
      int h = C >> 6, hk = C & (HK - 1);
      dst[(((size_t)(b * NHh + h)) * Nn + n) * HK + hk] = (_Float16)acc[c][v];
    }
  }
}

// ---- Kernel 2: conventional branch, flash attention ------------------------
// Grid: (32 bh, 16 q-tiles of 128), block 256; wave -> 16 query rows.
// K tile staged with async global->LDS B128 copies (overlapped with the
// VALU-transposing V staging), completed with s_wait_asynccnt + barrier.
__global__ __launch_bounds__(256) void flash_attn(const _Float16* __restrict__ Qh,
                                                  const _Float16* __restrict__ Kh,
                                                  const _Float16* __restrict__ Vh,
                                                  _Float16* __restrict__ att) {
  __shared__ _Float16 Ktile[32 * 72];       // [key][hk], ld 72
  __shared__ _Float16 Vt[64 * 40];          // [hk][key], ld 40 (transposed)
  __shared__ _Float16 Pscr[8 * 16 * 40];    // per-wave P re-layout scratch

  const int tid = threadIdx.x, w = wave_id(), lane = lane_id();
  const int bh = blockIdx.x;                // b*8 + h
  const int b = bh >> 3, h = bh & 7;
  const int q0 = blockIdx.y * 128;
  const size_t rowbase = (size_t)bh * Nn;

  // Q fragments for this wave's 16 rows (hk 0..31 and 32..63)
  const _Float16* qp = Qh + (rowbase + q0 + w * 16) * HK;
  v16h qa0 = load_frag_a(qp, HK);
  v16h qa1 = load_frag_a(qp + 32, HK);

  v8f acc[4];
#pragma unroll
  for (int c = 0; c < 4; ++c) acc[c] = (v8f){};
  float mrow[8], lrow[8];
#pragma unroll
  for (int v = 0; v < 8; ++v) { mrow[v] = -1e30f; lrow[v] = 0.f; }

  _Float16* Pw = Pscr + w * 16 * 40;
  const int g = lane >> 4, nl = lane & 15;

  // per-thread staging coordinates
  const int kr = tid >> 3, kc = (tid & 7) * 8;        // K tile: 16B per lane
  const unsigned klds = (unsigned)(uintptr_t)&Ktile[kr * 72 + kc];
  const int vd = tid >> 2, vk = (tid & 3) * 8;        // V transpose gather

  for (int j = 0; j < Nn; j += 32) {
    __syncthreads();                       // prior-iteration LDS reads done
    // issue async K-tile copy, then overlap with the V transpose staging
    async_ld_b128(klds, &Kh[(rowbase + j + kr) * HK + kc]);
#pragma unroll
    for (int i = 0; i < 8; ++i)
      Vt[vd * 40 + vk + i] = Vh[(rowbase + j + vk + i) * HK + vd];
    wait_async0();
    __syncthreads();

    // S = Q Kt : two 16x16 score tiles (keys j..j+15, j+16..j+31)
    v8f s0 = (v8f){}, s1 = (v8f){};
    s0 = WMMA(qa0, load_frag_b_t(Ktile, 72), s0);
    s0 = WMMA(qa1, load_frag_b_t(Ktile + 32, 72), s0);
    s1 = WMMA(qa0, load_frag_b_t(Ktile + 16 * 72, 72), s1);
    s1 = WMMA(qa1, load_frag_b_t(Ktile + 16 * 72 + 32, 72), s1);

    float corr[8];
#pragma unroll
    for (int v = 0; v < 8; ++v) {
      float a0 = s0[v] * SCALE, a1 = s1[v] * SCALE;
      float mx = fmaxf(a0, a1);
#pragma unroll
      for (int msk = 1; msk < 16; msk <<= 1) mx = fmaxf(mx, __shfl_xor(mx, msk, 32));
      float mnew = fmaxf(mrow[v], mx);
      corr[v] = __expf(mrow[v] - mnew);
      mrow[v] = mnew;
      float p0 = __expf(a0 - mnew), p1 = __expf(a1 - mnew);
      float rs = p0 + p1;
#pragma unroll
      for (int msk = 1; msk < 16; msk <<= 1) rs += __shfl_xor(rs, msk, 32);
      lrow[v] = lrow[v] * corr[v] + rs;
      s0[v] = p0; s1[v] = p1;
    }
#pragma unroll
    for (int c = 0; c < 4; ++c)
#pragma unroll
      for (int v = 0; v < 8; ++v) acc[c][v] *= corr[v];

    // C-frag -> A-frag re-layout of P through LDS
#pragma unroll
    for (int v = 0; v < 8; ++v) {
      Pw[(v + 8 * g) * 40 + nl]      = (_Float16)s0[v];
      Pw[(v + 8 * g) * 40 + 16 + nl] = (_Float16)s1[v];
    }
    __syncthreads();

    v16h pa = load_frag_a(Pw, 40);
#pragma unroll
    for (int c = 0; c < 4; ++c) {
      v16h vb = load_frag_b_t(Vt + c * 16 * 40, 40);
      acc[c] = WMMA(pa, vb, acc[c]);
    }
  }

  // normalize and store att_c (f16) at cols h*64 + hk of the concat buffer
#pragma unroll
  for (int v = 0; v < 8; ++v) {
    float inv = 1.0f / lrow[v];
    int n = q0 + w * 16 + v + 8 * g;
#pragma unroll
    for (int c = 0; c < 4; ++c) {
      int C = h * HK + c * 16 + nl;
      att[((size_t)(b * Nn + n)) * (2 * Dd) + C] = (_Float16)(acc[c][v] * inv);
    }
  }
}

// ---- Kernel 3: feature branch scores: softmax(Kt Q * scale) ----------------
// One block per (b,h). Result stored transposed: distT[q][d], f16.
__global__ __launch_bounds__(256) void feat_scores(const _Float16* __restrict__ Kh,
                                                   const _Float16* __restrict__ Qh,
                                                   _Float16* __restrict__ distT) {
  __shared__ _Float16 Kt[64 * 40];          // [hk_k][n'], transposed
  __shared__ _Float16 Qt[64 * 40];          // [hk_q][n'], transposed
  __shared__ float Sf[64 * 65];

  const int tid = threadIdx.x, w = wave_id(), lane = lane_id();
  const int bh = blockIdx.x;
  const size_t rowbase = (size_t)bh * Nn;
  const int ti = w >> 1;                    // row tile of Sf (hk_k)
  const int tj0 = (w & 1) * 2;              // two col tiles (hk_q)

  v8f acc0 = (v8f){}, acc1 = (v8f){};

  for (int nb = 0; nb < Nn; nb += 32) {
    __syncthreads();
    {
      int d = tid >> 2, n0 = (tid & 3) * 8;
#pragma unroll
      for (int i = 0; i < 8; ++i) {
        Kt[d * 40 + n0 + i] = Kh[(rowbase + nb + n0 + i) * HK + d];
        Qt[d * 40 + n0 + i] = Qh[(rowbase + nb + n0 + i) * HK + d];
      }
    }
    __syncthreads();
    v16h a = load_frag_a(Kt + ti * 16 * 40, 40);
    acc0 = WMMA(a, load_frag_b_t(Qt + tj0 * 16 * 40, 40), acc0);
    acc1 = WMMA(a, load_frag_b_t(Qt + (tj0 + 1) * 16 * 40, 40), acc1);
  }

  const int g = lane >> 4, nl = lane & 15;
#pragma unroll
  for (int v = 0; v < 8; ++v) {
    int r = ti * 16 + v + 8 * g;
    Sf[r * 65 + tj0 * 16 + nl]       = acc0[v] * SCALE;
    Sf[r * 65 + (tj0 + 1) * 16 + nl] = acc1[v] * SCALE;
  }
  __syncthreads();

  if (tid < 64) {
    int i = tid;
    float mx = -1e30f;
    for (int j = 0; j < 64; ++j) mx = fmaxf(mx, Sf[i * 65 + j]);
    float sum = 0.f;
    for (int j = 0; j < 64; ++j) {
      float e = __expf(Sf[i * 65 + j] - mx);
      Sf[i * 65 + j] = e;
      sum += e;
    }
    float inv = 1.0f / sum;
    for (int j = 0; j < 64; ++j)
      distT[(size_t)bh * 4096 + j * 64 + i] = (_Float16)(Sf[i * 65 + j] * inv);
  }
}

// ---- Kernel 4: att_f = V @ dist  --------------------------------------------
// Grid: (32 bh, 16 tiles of 128 rows); direct-from-global fragments.
__global__ __launch_bounds__(256) void feat_apply(const _Float16* __restrict__ Vh,
                                                  const _Float16* __restrict__ distT,
                                                  _Float16* __restrict__ att) {
  const int w = wave_id(), lane = lane_id();
  const int bh = blockIdx.x;
  const int b = bh >> 3, h = bh & 7;
  const int row0 = blockIdx.y * 128 + w * 16;
  const _Float16* vp = Vh + ((size_t)bh * Nn + row0) * HK;
  const _Float16* dp = distT + (size_t)bh * 4096;

  v16h a0 = load_frag_a(vp, HK);
  v16h a1 = load_frag_a(vp + 32, HK);

  const int g = lane >> 4, nl = lane & 15;
#pragma unroll
  for (int c = 0; c < 4; ++c) {
    v8f acc = (v8f){};
    acc = WMMA(a0, load_frag_b_t(dp + c * 16 * 64, 64), acc);
    acc = WMMA(a1, load_frag_b_t(dp + c * 16 * 64 + 32, 64), acc);
#pragma unroll
    for (int v = 0; v < 8; ++v) {
      int n = row0 + v + 8 * g;
      int C = Dd + h * HK + c * 16 + nl;   // second half of concat
      att[((size_t)(b * Nn + n)) * (2 * Dd) + C] = (_Float16)acc[v];
    }
  }
}

// ---- Kernel 5: FC gemm + bias + relu + residual -> tmp f32 -----------------
// att[8192,1024] f16 @ Wf[1024,512] f32 ; grid (8192/128, 512/64), block 256.
__global__ __launch_bounds__(256) void fc_gemm(const _Float16* __restrict__ att,
                                               const float* __restrict__ Wf,
                                               const float* __restrict__ bf,
                                               const float* __restrict__ X,
                                               float* __restrict__ tmp) {
  __shared__ _Float16 Wt[64 * 36];          // [col][k], transposed
  const int tid = threadIdx.x, w = wave_id(), lane = lane_id();
  const int row0 = blockIdx.x * 128;
  const int cb   = blockIdx.y * 64;

  v8f acc[4];
#pragma unroll
  for (int c = 0; c < 4; ++c) acc[c] = (v8f){};

  for (int kb = 0; kb < 2 * Dd; kb += 32) {
    __syncthreads();
    {
      int c = tid >> 2, k0 = (tid & 3) * 8;
#pragma unroll
      for (int i = 0; i < 8; ++i)
        Wt[c * 36 + k0 + i] = (_Float16)Wf[(size_t)(kb + k0 + i) * Dd + cb + c];
      if (kb + 32 < 2 * Dd)  // prefetch next weight chunk
        __builtin_prefetch(&Wf[(size_t)(kb + 32 + k0) * Dd + cb + c], 0, 1);
    }
    __syncthreads();
    v16h a = load_frag_a(att + (size_t)(row0 + w * 16) * (2 * Dd) + kb, 2 * Dd);
#pragma unroll
    for (int c = 0; c < 4; ++c)
      acc[c] = WMMA(a, load_frag_b_t(Wt + c * 16 * 36, 36), acc[c]);
  }

  const int g = lane >> 4, nl = lane & 15;
#pragma unroll
  for (int c = 0; c < 4; ++c)
#pragma unroll
    for (int v = 0; v < 8; ++v) {
      int R = row0 + w * 16 + v + 8 * g;
      int C = cb + c * 16 + nl;
      float val = acc[c][v] + bf[C];
      val = fmaxf(val, 0.f) + X[(size_t)R * Dd + C];
      tmp[(size_t)R * Dd + C] = val;
    }
}

// ---- Kernel 6: LayerNorm ----------------------------------------------------
__global__ __launch_bounds__(256) void layernorm(const float* __restrict__ tmp,
                                                 const float* __restrict__ gamma,
                                                 const float* __restrict__ beta,
                                                 float* __restrict__ out) {
  __shared__ float ssum[256], ssq[256];
  const int r = blockIdx.x, tid = threadIdx.x;
  const float v0 = tmp[(size_t)r * Dd + tid];
  const float v1 = tmp[(size_t)r * Dd + 256 + tid];
  ssum[tid] = v0 + v1;
  ssq[tid]  = v0 * v0 + v1 * v1;
  __syncthreads();
  for (int s = 128; s > 0; s >>= 1) {
    if (tid < s) { ssum[tid] += ssum[tid + s]; ssq[tid] += ssq[tid + s]; }
    __syncthreads();
  }
  const float mean = ssum[0] * (1.0f / Dd);
  const float var  = ssq[0] * (1.0f / Dd) - mean * mean;
  const float rstd = __frsqrt_rn(var + LN_EPS);
  out[(size_t)r * Dd + tid]       = (v0 - mean) * rstd * gamma[tid] + beta[tid];
  out[(size_t)r * Dd + 256 + tid] = (v1 - mean) * rstd * gamma[256 + tid] + beta[256 + tid];
}

// ---- Host side --------------------------------------------------------------
extern "C" void kernel_launch(void* const* d_in, const int* in_sizes, int n_in,
                              void* d_out, int out_size, void* d_ws, size_t ws_size,
                              hipStream_t stream) {
  (void)in_sizes; (void)n_in; (void)out_size; (void)ws_size;
  const float* x     = (const float*)d_in[0];
  const float* Wq    = (const float*)d_in[1];
  const float* Wk    = (const float*)d_in[2];
  const float* Wv    = (const float*)d_in[3];
  const float* Wf    = (const float*)d_in[4];
  const float* bf    = (const float*)d_in[5];
  const float* gamma = (const float*)d_in[6];
  const float* beta  = (const float*)d_in[7];
  float* out = (float*)d_out;

  const size_t QKVe = (size_t)Bz * NHh * Nn * HK;        // 4,194,304 f16
  _Float16* Qh    = (_Float16*)d_ws;
  _Float16* Kh    = Qh + QKVe;
  _Float16* Vh    = Kh + QKVe;
  _Float16* att   = Vh + QKVe;                           // [8192][1024] f16
  _Float16* distT = att + (size_t)Mrows * 2 * Dd;        // 32*64*64 f16
  float*    tmp   = (float*)(distT + 32 * 4096);         // [8192][512] f32

  dim3 blk(256);
  proj_gemm<<<dim3(Mrows / 128, Dd / 64), blk, 0, stream>>>(x, Wq, Qh);
  proj_gemm<<<dim3(Mrows / 128, Dd / 64), blk, 0, stream>>>(x, Wk, Kh);
  proj_gemm<<<dim3(Mrows / 128, Dd / 64), blk, 0, stream>>>(x, Wv, Vh);
  flash_attn<<<dim3(Bz * NHh, Nn / 128), blk, 0, stream>>>(Qh, Kh, Vh, att);
  feat_scores<<<dim3(Bz * NHh), blk, 0, stream>>>(Kh, Qh, distT);
  feat_apply<<<dim3(Bz * NHh, Nn / 128), blk, 0, stream>>>(Vh, distT, att);
  fc_gemm<<<dim3(Mrows / 128, Dd / 64), blk, 0, stream>>>(att, Wf, bf, x, tmp);
  layernorm<<<dim3(Mrows), blk, 0, stream>>>(tmp, gamma, beta, out);
}
